// GraphAttentionLayer_50946902065864
// MI455X (gfx1250) — matrix operand order repaired and checked by reading, not compiled, not run on hardware
//
#include <hip/hip_runtime.h>
#include <hip/hip_bf16.h>
#include <stdint.h>

typedef __bf16 bf16;
typedef __attribute__((ext_vector_type(16))) __bf16 v16bf;
typedef __attribute__((ext_vector_type(8)))  float  v8f;
typedef unsigned int u32x4 __attribute__((ext_vector_type(4)));
typedef int          i32x8 __attribute__((ext_vector_type(8)));
typedef int          i32x4 __attribute__((ext_vector_type(4)));

// ---------- constants ----------
#define Bc 8
#define Nc 1024
#define Dc 512
#define Hc 8
#define Ec 8
#define HDc 64

union Frag {
    v16bf v;
    uint4 u[2];
};

__device__ __forceinline__ bf16 to_bf16(float f) {
    unsigned u = __builtin_bit_cast(unsigned, f);
    unsigned r = (u + 0x7FFFu + ((u >> 16) & 1u)) >> 16;   // RNE
    unsigned short h = (unsigned short)r;
    return __builtin_bit_cast(bf16, h);
}

__device__ __forceinline__ v8f bwmma(v16bf a, v16bf b, v8f c) {
    return __builtin_amdgcn_wmma_f32_16x16x32_bf16(
        false, a, false, b, (short)0, c, false, false);
}

// A fragment (16x32 bf16, M = lane%16): lanes 0-15 hold K 0..15,
// lanes 16-31 hold K 16..31.
__device__ __forceinline__ v16bf load_a_frag(const bf16* row, int lane) {
    int kb = (lane & 16) ? 16 : 0;
    Frag f;
    f.u[0] = *(const uint4*)(row + kb);
    f.u[1] = *(const uint4*)(row + kb + 8);
    return f.v;
}

// B fragment (32x16 bf16, N = lane%16): each lane = 16 contiguous K elements.
__device__ __forceinline__ v16bf load_b_frag(const bf16* p) {
    Frag f;
    f.u[0] = *(const uint4*)(p);
    f.u[1] = *(const uint4*)(p + 8);
    return f.v;
}

// ---------- Tensor Data Mover: 2D tile -> LDS (D# per ISA §8.3/8.4) ----------
// data_size_code: 0=1B 1=2B 2=4B. Dims/strides in data_size units.
__device__ __forceinline__ void tdm_load_2d(unsigned lds_addr, const void* gaddr,
                                            unsigned data_size_code,
                                            unsigned tile_d0, unsigned tile_d1,
                                            unsigned long long stride0,
                                            unsigned tensor_d0, unsigned tensor_d1) {
    unsigned long long ga = (unsigned long long)(uintptr_t)gaddr;
    u32x4 g0;
    g0[0] = 1u;                                            // count=1, user mode
    g0[1] = lds_addr;                                      // LDS byte address
    g0[2] = (unsigned)(ga & 0xFFFFFFFFull);                // global_addr[31:0]
    g0[3] = (unsigned)((ga >> 32) & 0x01FFFFFFull)         // global_addr[56:32]
            | (2u << 30);                                  // type=2 (image)
    i32x8 g1;
    g1[0] = (int)(data_size_code << 16);                   // mask=0, flags=0
    g1[1] = (int)((tensor_d0 & 0xFFFFu) << 16);            // tensor_dim0[15:0]
    g1[2] = (int)((tensor_d0 >> 16) | ((tensor_d1 & 0xFFFFu) << 16));
    g1[3] = (int)((tensor_d1 >> 16) | (tile_d0 << 16));    // tile_dim0
    g1[4] = (int)(tile_d1 & 0xFFFFu);                      // tile_dim1, tile_dim2=0
    g1[5] = (int)(stride0 & 0xFFFFFFFFull);                // tensor_dim0_stride[31:0]
    g1[6] = (int)((stride0 >> 32) & 0xFFFFull);            // stride0[47:32], stride1=0
    g1[7] = 0;
    i32x4 z4 = {0, 0, 0, 0};
    i32x8 z8 = {0, 0, 0, 0, 0, 0, 0, 0};
    __builtin_amdgcn_tensor_load_to_lds(g0, g1, z4, z4, z8, 0);
}

__device__ __forceinline__ unsigned lds_off(const void* p) {
    // LDS aperture: flat addr[31:0] == LDS byte offset
    return (unsigned)(uintptr_t)p;
}

// ---------- conversion kernels ----------
__global__ __launch_bounds__(256) void cvt_bf16_kernel(const float* __restrict__ s,
                                                       bf16* __restrict__ d, int n) {
    int i = blockIdx.x * 256 + threadIdx.x;
    if (i < n) d[i] = to_bf16(s[i]);
}

__global__ __launch_bounds__(256) void cvt_t_bf16_kernel(const float* __restrict__ s,
                                                         bf16* __restrict__ d, int R, int C) {
    int i = blockIdx.x * 256 + threadIdx.x;
    if (i < R * C) {
        int r = i / C, c = i % C;
        d[(size_t)c * R + r] = to_bf16(s[i]);
    }
}

// ---------- GEMM mainloop: 16x64 tile per wave; K = Dc (compile-time) ----------
// Simple form: compiler unrolls, clauses the b128 loads ahead of the WMMAs.
__device__ __forceinline__ void gemm_mainloop(const bf16* __restrict__ arow,
                                              const bf16* __restrict__ BT,
                                              int n0, int lane, v8f acc[4]) {
    int li = lane & 15;
    int khalf = (lane & 16) ? 16 : 0;
#pragma unroll 4
    for (int k0 = 0; k0 < Dc; k0 += 32) {
        v16bf a = load_a_frag(arow + k0, lane);
#pragma unroll
        for (int nc = 0; nc < 4; ++nc) {
            const bf16* bp = BT + (size_t)(n0 + nc * 16 + li) * Dc + k0 + khalf;
            v16bf b = load_b_frag(bp);
            acc[nc] = bwmma(a, b, acc[nc]);
        }
    }
}

// ---------- kernel 1: qkv = x @ Wqkv + b, scattered to Q/K/Vt bf16 ----------
__global__ __launch_bounds__(256) void qkv_gemm_kernel(const bf16* __restrict__ A,
                                                       const bf16* __restrict__ BT,
                                                       const float* __restrict__ bias,
                                                       bf16* __restrict__ Q,
                                                       bf16* __restrict__ K,
                                                       bf16* __restrict__ Vt) {
    int lane = threadIdx.x & 31;
    int wave = threadIdx.x >> 5;
    int li = lane & 15, lh = lane >> 4;
    int m0 = blockIdx.x * 128 + wave * 16;
    int n0 = blockIdx.y * 64;
    const bf16* arow = A + (size_t)(m0 + li) * Dc;
    v8f acc[4] = {};
    gemm_mainloop(arow, BT, n0, lane, acc);
#pragma unroll
    for (int nc = 0; nc < 4; ++nc) {
        int c = n0 + nc * 16 + li;
        float bc = bias[c];
        int which = c >> 9;
        int h = (c >> 6) & 7;
        int d = c & 63;
#pragma unroll
        for (int r = 0; r < 8; ++r) {
            int m = m0 + r + 8 * lh;
            int b = m >> 10, n = m & 1023;
            bf16 bv = to_bf16(acc[nc][r] + bc);
            size_t bh = (size_t)(b * Hc + h);
            if (which == 0)      Q[(bh * Nc + n) * HDc + d] = bv;
            else if (which == 1) K[(bh * Nc + n) * HDc + d] = bv;
            else                 Vt[(bh * HDc + d) * Nc + n] = bv;
        }
    }
}

// ---------- kernel 2: fused graph attention, TDM double-buffered tiles ----------
__global__ __launch_bounds__(256) void attn_kernel(const bf16* __restrict__ Q,
                                                   const bf16* __restrict__ K,
                                                   const bf16* __restrict__ Vt,
                                                   const float* __restrict__ ef,
                                                   const unsigned char* __restrict__ adj,
                                                   const unsigned char* __restrict__ nmask,
                                                   const float* __restrict__ W_eb,
                                                   const float* __restrict__ b_eb,
                                                   bf16* __restrict__ O) {
    __shared__ __align__(16) float s_ef[2][16 * 32 * Ec];      // 2 x 16 KB
    __shared__ __align__(16) unsigned char s_adj[2][16 * 32];  // 2 x 512 B
    __shared__ __align__(16) unsigned char s_nmj[2][32];
    __shared__ __align__(16) bf16 s_p[Hc][16 * 32];            // 8 KB

    int b = blockIdx.x;
    int i0 = blockIdx.y * 16;
    int wave = threadIdx.x >> 5;
    int lane = threadIdx.x & 31;
    int h = wave;
    int li = lane & 15, lh = lane >> 4;

    const float* ef_base = ef + ((size_t)b * Nc + i0) * Nc * Ec;
    const unsigned char* adj_base = adj + ((size_t)b * Nc + i0) * Nc;
    const unsigned char* nm_base = nmask + (size_t)b * Nc;

    // issue the 3 TDM descriptors for j-tile j0 into buffer `buf` (wave-level op)
    auto tdm_issue = [&](int j0, int buf) {
        tdm_load_2d(lds_off(&s_ef[buf][0]), ef_base + (size_t)j0 * Ec,
                    /*4B*/ 2, /*tile*/ 32 * Ec, 16,
                    /*stride0*/ (unsigned long long)Nc * Ec,
                    /*tensor*/ Nc * Ec, Nc);
        tdm_load_2d(lds_off(&s_adj[buf][0]), adj_base + j0,
                    /*1B*/ 0, /*tile*/ 32, 16,
                    /*stride0*/ Nc, /*tensor*/ Nc, Nc);
        tdm_load_2d(lds_off(&s_nmj[buf][0]), nm_base + j0,
                    /*1B*/ 0, /*tile*/ 32, 1,
                    /*stride0*/ Nc, /*tensor*/ Nc, 1);
    };

    if (wave == 0) tdm_issue(0, 0);   // prefetch first tile

    float web[Ec];
#pragma unroll
    for (int e = 0; e < Ec; ++e) web[e] = W_eb[e * Hc + h];
    float beb = b_eb[h];

    const bf16* Qb = Q + ((size_t)(b * Hc + h) * Nc + i0) * HDc;
    v16bf aq0 = load_a_frag(Qb + (size_t)li * HDc + 0, lane);
    v16bf aq1 = load_a_frag(Qb + (size_t)li * HDc + 32, lane);

    float m_run[8], l_run[8];
    unsigned char nm_i[8];
    v8f o_acc[4] = {};
#pragma unroll
    for (int r = 0; r < 8; ++r) {
        m_run[r] = -1e30f;
        l_run[r] = 0.0f;
        nm_i[r] = nm_base[i0 + r + 8 * lh];
    }

    for (int j0 = 0; j0 < Nc; j0 += 32) {
        int buf = (j0 >> 5) & 1;
        bool has_next = (j0 + 32) < Nc;
        if (wave == 0 && has_next) tdm_issue(j0 + 32, buf ^ 1);

        // scores (LDS-independent): overlap with TDM of current+next tiles
        v8f s[2];
#pragma unroll
        for (int js = 0; js < 2; ++js) {
            const bf16* Kb = K + ((size_t)(b * Hc + h) * Nc + j0 + js * 16 + li) * HDc;
            int khalf = lh ? 16 : 0;
            v16bf bk0 = load_b_frag(Kb + khalf);
            v16bf bk1 = load_b_frag(Kb + 32 + khalf);
            v8f z = {};
            z = bwmma(aq0, bk0, z);
            z = bwmma(aq1, bk1, z);
            s[js] = z;
        }

        if (wave == 0) {
            if (has_next) __builtin_amdgcn_s_wait_tensorcnt(3);  // current tile landed
            else          __builtin_amdgcn_s_wait_tensorcnt(0);
        }
        __syncthreads();   // all waves: current tile visible in LDS

        // scale + edge bias + mask, element (i = r+8*lh, j = j0 + js*16 + li)
#pragma unroll
        for (int js = 0; js < 2; ++js) {
#pragma unroll
            for (int r = 0; r < 8; ++r) {
                int i = r + 8 * lh;
                int jl = js * 16 + li;
                const float* e8 = &s_ef[buf][(i * 32 + jl) * Ec];
                float bias = beb;
#pragma unroll
                for (int e = 0; e < Ec; ++e) bias = fmaf(e8[e], web[e], bias);
                bool msk = s_adj[buf][i * 32 + jl] && nm_i[r] && s_nmj[buf][jl];
                s[js][r] = msk ? fmaf(s[js][r], 0.125f, bias) : -1e9f;
            }
        }

        // online softmax (row reductions within 16-lane halves)
#pragma unroll
        for (int r = 0; r < 8; ++r) {
            float mx = fmaxf(s[0][r], s[1][r]);
#pragma unroll
            for (int off = 1; off < 16; off <<= 1)
                mx = fmaxf(mx, __shfl_xor(mx, off, 32));
            float mnew = fmaxf(m_run[r], mx);
            float alpha = __expf(m_run[r] - mnew);
            m_run[r] = mnew;
            float p0 = __expf(s[0][r] - mnew);
            float p1 = __expf(s[1][r] - mnew);
            float rs = p0 + p1;
#pragma unroll
            for (int off = 1; off < 16; off <<= 1)
                rs += __shfl_xor(rs, off, 32);
            l_run[r] = l_run[r] * alpha + rs;
            s[0][r] = p0;
            s[1][r] = p1;
#pragma unroll
            for (int dc = 0; dc < 4; ++dc) o_acc[dc][r] *= alpha;
        }

        // P (D-layout) -> LDS -> A-fragment relayout (same-wave DS in-order)
        bf16* pw = &s_p[wave][0];
#pragma unroll
        for (int js = 0; js < 2; ++js)
#pragma unroll
            for (int r = 0; r < 8; ++r)
                pw[(r + 8 * lh) * 32 + js * 16 + li] = to_bf16(s[js][r]);
        asm volatile("s_wait_dscnt 0x0" ::: "memory");
        v16bf ap;
        {
            int kb = lh ? 16 : 0;
            const bf16* prow = pw + li * 32;
            Frag f;
            f.u[0] = *(const uint4*)(prow + kb);
            f.u[1] = *(const uint4*)(prow + kb + 8);
            ap = f.v;
        }

        // O += P @ V
#pragma unroll
        for (int dc = 0; dc < 4; ++dc) {
            const bf16* Vb = Vt + ((size_t)(b * Hc + h) * HDc + dc * 16 + li) * Nc
                             + j0 + (lh ? 16 : 0);
            v16bf bv = load_b_frag(Vb);
            o_acc[dc] = bwmma(ap, bv, o_acc[dc]);
        }

        __syncthreads();   // reads of `buf` done before a later issue overwrites it
    }

    // epilogue: normalize, store bf16 O[B*N, D]
#pragma unroll
    for (int dc = 0; dc < 4; ++dc) {
#pragma unroll
        for (int r = 0; r < 8; ++r) {
            int i = r + 8 * lh;
            float val = o_acc[dc][r] / l_run[r];
            O[((size_t)(b * Nc) + i0 + i) * Dc + h * HDc + dc * 16 + li] = to_bf16(val);
        }
    }
}

// ---------- kernel 3: final projection, f32 out + bias ----------
__global__ __launch_bounds__(256) void out_gemm_kernel(const bf16* __restrict__ A,
                                                       const bf16* __restrict__ BT,
                                                       const float* __restrict__ bias,
                                                       float* __restrict__ out) {
    int lane = threadIdx.x & 31;
    int wave = threadIdx.x >> 5;
    int li = lane & 15, lh = lane >> 4;
    int m0 = blockIdx.x * 128 + wave * 16;
    int n0 = blockIdx.y * 64;
    const bf16* arow = A + (size_t)(m0 + li) * Dc;
    v8f acc[4] = {};
    gemm_mainloop(arow, BT, n0, lane, acc);
#pragma unroll
    for (int nc = 0; nc < 4; ++nc) {
        int c = n0 + nc * 16 + li;
        float bc = bias[c];
#pragma unroll
        for (int r = 0; r < 8; ++r) {
            int m = m0 + r + 8 * lh;
            out[(size_t)m * Dc + c] = acc[nc][r] + bc;
        }
    }
}

// ---------- host launcher ----------
extern "C" void kernel_launch(void* const* d_in, const int* in_sizes, int n_in,
                              void* d_out, int out_size, void* d_ws, size_t ws_size,
                              hipStream_t stream) {
    const float* x      = (const float*)d_in[0];
    const unsigned char* adj   = (const unsigned char*)d_in[1];   // bool [B,N,N]
    const unsigned char* nmask = (const unsigned char*)d_in[2];   // bool [B,N]
    const float* ef     = (const float*)d_in[3];
    const float* W_qkv  = (const float*)d_in[4];
    const float* b_qkv  = (const float*)d_in[5];
    const float* W_eb   = (const float*)d_in[6];
    const float* b_eb   = (const float*)d_in[7];
    const float* W_out  = (const float*)d_in[8];
    const float* b_out  = (const float*)d_in[9];
    float* out = (float*)d_out;

    char* ws = (char*)d_ws;
    bf16* xb    = (bf16*)(ws + 0);                 //  8 MB  x bf16 [8192,512]
    bf16* WqkvT = (bf16*)(ws + 8388608);           //  1.5MB Wqkv^T [1536,512]
    bf16* WoutT = (bf16*)(ws + 9961472);           //  0.5MB Wout^T [512,512]
    bf16* Q     = (bf16*)(ws + 10485760);          //  8 MB  [B,H,N,64]
    bf16* K     = (bf16*)(ws + 18874368);          //  8 MB  [B,H,N,64]
    bf16* Vt    = (bf16*)(ws + 27262976);          //  8 MB  [B,H,64,N]
    bf16* O     = (bf16*)(ws + 35651584);          //  8 MB  attn out [8192,512]

    {
        int n = Bc * Nc * Dc;
        cvt_bf16_kernel<<<(n + 255) / 256, 256, 0, stream>>>(x, xb, n);
    }
    {
        int n = Dc * 3 * Dc;
        cvt_t_bf16_kernel<<<(n + 255) / 256, 256, 0, stream>>>(W_qkv, WqkvT, Dc, 3 * Dc);
    }
    {
        int n = Dc * Dc;
        cvt_t_bf16_kernel<<<(n + 255) / 256, 256, 0, stream>>>(W_out, WoutT, Dc, Dc);
    }

    qkv_gemm_kernel<<<dim3(Bc * Nc / 128, 3 * Dc / 64), 256, 0, stream>>>(
        xb, WqkvT, b_qkv, Q, K, Vt);

    attn_kernel<<<dim3(Bc, Nc / 16), 256, 0, stream>>>(
        Q, K, Vt, ef, adj, nmask, W_eb, b_eb, O);

    out_gemm_kernel<<<dim3(Bc * Nc / 128, Dc / 64), 256, 0, stream>>>(
        O, WoutT, b_out, out);
}
